// SoftCLDiceLoss_86440511799616
// MI455X (gfx1250) — compile-verified, matrix operand order checked by reading, not run on hardware
//
#include <hip/hip_runtime.h>
#include <cstdint>
#include <cstddef>

// ---------------------------------------------------------------------------
// Geometry constants
// ---------------------------------------------------------------------------
constexpr int kD     = 96;
constexpr int kH     = 192;
constexpr int kW     = 192;
constexpr int kWPR   = 6;                 // 192 / 32 words per row (exact)
constexpr int kSlice = kH * kWPR;         // 1152 words per slice
constexpr int kNWB   = kD * kH * kWPR;    // 110592 words per (batch) volume
constexpr int kNW    = 4 * kNWB;          // 442368: [x_all b0, x_all b1, y_all b0, y_all b1]
constexpr int kNRow  = kNW / kWPR;        // 73728 packed rows total
constexpr int kRowsPerBatch = kD * kH;    // 18432
constexpr int kCh    = kD * kH * kW;      // 3538944 floats channel stride
constexpr int kIters = 10;

typedef __attribute__((ext_vector_type(16))) _Float16 v16h;
typedef __attribute__((ext_vector_type(8)))  float    v8f;

// ---------------------------------------------------------------------------
// Row helpers: a packed W-row is 6 words = 24 bytes, 8-byte aligned.
// ---------------------------------------------------------------------------
__device__ __forceinline__ void load_row(const uint32_t* __restrict__ p, uint32_t r[6]) {
  const uint2* q = (const uint2*)p;
  uint2 a = q[0], b = q[1], c = q[2];
  r[0] = a.x; r[1] = a.y; r[2] = b.x; r[3] = b.y; r[4] = c.x; r[5] = c.y;
}
__device__ __forceinline__ void or_row(const uint32_t* __restrict__ p, uint32_t r[6]) {
  const uint2* q = (const uint2*)p;
  uint2 a = q[0], b = q[1], c = q[2];
  r[0] |= a.x; r[1] |= a.y; r[2] |= b.x; r[3] |= b.y; r[4] |= c.x; r[5] |= c.y;
}
__device__ __forceinline__ void store_row(uint32_t* __restrict__ p, const uint32_t r[6]) {
  uint2* q = (uint2*)p;
  q[0] = make_uint2(r[0], r[1]);
  q[1] = make_uint2(r[2], r[3]);
  q[2] = make_uint2(r[4], r[5]);
}

// ---------------------------------------------------------------------------
// 1) Binarize + bit-pack.  word i covers 32 voxels along W.
//    b' in [0,2): x_all = (x[b,1,...] > x[b,0,...]); b' in [2,4): y_all = (y>0)
// ---------------------------------------------------------------------------
__global__ void k_binarize(const float* __restrict__ x, const float* __restrict__ y,
                           uint32_t* __restrict__ bits, uint32_t* __restrict__ cur) {
  int i = blockIdx.x * blockDim.x + threadIdx.x;
  if (i >= kNW) return;
  int b  = i / kNWB;
  int r  = i - b * kNWB;
  int d  = r / kSlice;
  int r2 = r - d * kSlice;
  int h  = r2 / kWPR;
  int w0 = (r2 - h * kWPR) * 32;

  uint32_t bv = 0u;
  if (b < 2) {
    const float* p0 = x + (size_t)(b * 2 + 0) * kD * (size_t)(kH * kW)
                        + (size_t)d * (kH * kW) + (size_t)h * kW + w0;
    const float* p1 = p0 + (size_t)kCh;   // channel 1
    #pragma unroll 8
    for (int j = 0; j < 32; ++j)
      if (p1[j] > p0[j]) bv |= (1u << j);
  } else {
    const float* py = y + (size_t)(b - 2) * kD * (size_t)(kH * kW)
                        + (size_t)d * (kH * kW) + (size_t)h * kW + w0;
    #pragma unroll 8
    for (int j = 0; j < 32; ++j)
      if (py[j] > 0.0f) bv |= (1u << j);
  }
  bits[i] = bv;
  cur[i]  = bv;
}

// ---------------------------------------------------------------------------
// 2) Cross-SE erosion, one packed W-row (6 words) per thread:
//    AND over {c, w±1, h±1, d±1}; out-of-range == 1 (+inf min padding).
//    W-arm is pure in-register funnel shifting.
// ---------------------------------------------------------------------------
__global__ void k_erode_row(const uint32_t* __restrict__ in, uint32_t* __restrict__ out) {
  int row = blockIdx.x * blockDim.x + threadIdx.x;
  if (row >= kNRow) return;
  int rb = row % kRowsPerBatch;
  int d  = rb / kH;
  int h  = rb - d * kH;

  const uint32_t* base = in + (size_t)row * kWPR;
  uint32_t c[6], up[6], dn[6], fr[6], bk[6];
  load_row(base, c);
  if (h > 0)      load_row(base - kWPR,   up); else { for (int j = 0; j < 6; ++j) up[j] = 0xFFFFFFFFu; }
  if (h < kH - 1) load_row(base + kWPR,   dn); else { for (int j = 0; j < 6; ++j) dn[j] = 0xFFFFFFFFu; }
  if (d > 0)      load_row(base - kSlice, fr); else { for (int j = 0; j < 6; ++j) fr[j] = 0xFFFFFFFFu; }
  if (d < kD - 1) load_row(base + kSlice, bk); else { for (int j = 0; j < 6; ++j) bk[j] = 0xFFFFFFFFu; }

  uint32_t o[6];
  #pragma unroll
  for (int j = 0; j < 6; ++j) {
    uint32_t sm = (c[j] << 1) | ((j > 0) ? (c[j - 1] >> 31) : 1u);          // w-1 (global w==0 -> 1)
    uint32_t sp = (c[j] >> 1) | ((j < 5) ? (c[j + 1] << 31) : 0x80000000u); // w+1 (global w==191 -> 1)
    o[j] = c[j] & sm & sp & up[j] & dn[j] & fr[j] & bk[j];
  }
  store_row(out + (size_t)row * kWPR, o);
}

// ---------------------------------------------------------------------------
// 3) Fused 3x3x3 dilation of `in` + skeleton update, one row per thread:
//    V = OR of up to 9 rows; open = V | shift(V,±1) (missing -> 0);
//    sv = xc & ~open; skel = set ? sv : (skel | sv)
// ---------------------------------------------------------------------------
__global__ void k_dilate_skel_row(const uint32_t* __restrict__ in,
                                  const uint32_t* __restrict__ xc,
                                  uint32_t* __restrict__ skel, int setMode) {
  int row = blockIdx.x * blockDim.x + threadIdx.x;
  if (row >= kNRow) return;
  int rb = row % kRowsPerBatch;
  int d  = rb / kH;
  int h  = rb - d * kH;

  const uint32_t* base = in + (size_t)row * kWPR;
  uint32_t V[6] = {0u, 0u, 0u, 0u, 0u, 0u};
  #pragma unroll
  for (int dd = -1; dd <= 1; ++dd) {
    int d2 = d + dd;
    if (d2 < 0 || d2 >= kD) continue;
    #pragma unroll
    for (int hh = -1; hh <= 1; ++hh) {
      int h2 = h + hh;
      if (h2 < 0 || h2 >= kH) continue;
      or_row(base + (ptrdiff_t)dd * kSlice + (ptrdiff_t)hh * kWPR, V);
    }
  }
  uint32_t xr[6];
  load_row(xc + (size_t)row * kWPR, xr);

  uint32_t sv[6];
  #pragma unroll
  for (int j = 0; j < 6; ++j) {
    uint32_t open = V[j] | (V[j] << 1) | (V[j] >> 1);
    if (j > 0) open |= (V[j - 1] >> 31);
    if (j < 5) open |= (V[j + 1] << 31);
    sv[j] = xr[j] & ~open;
  }
  uint32_t* sp = skel + (size_t)row * kWPR;
  if (!setMode) {
    uint32_t sr[6];
    load_row(sp, sr);
    #pragma unroll
    for (int j = 0; j < 6; ++j) sv[j] |= sr[j];
  }
  store_row(sp, sv);
}

// ---------------------------------------------------------------------------
// 4) Popcount reductions via WMMA (v_wmma_f32_16x16x32_f16).
//    One wave handles a 512-word chunk: async-stage skel+mask chunk into LDS
//    (ASYNCcnt), per-lane popcounts (<=32, exact f16) -> A(16x32), B = ones,
//    accumulate in f32 C (exact: total <= 7.1M < 2^24).
//    Every column of C is identical, so total = laneSum + laneSum(lane^16).
//    acc[0]=sum(pred&y) acc[1]=sum(pred) acc[2]=sum(gt&x) acc[3]=sum(gt)
// ---------------------------------------------------------------------------
__global__ void k_reduce(const uint32_t* __restrict__ skel,
                         const uint32_t* __restrict__ bits,
                         float* __restrict__ acc) {
  __shared__ uint32_t smem[8][1024];      // per-wave: 512 skel + 512 mask words

  const int lane   = threadIdx.x & 31;
  const int wib    = threadIdx.x >> 5;    // wave in block (blockDim.x == 256)
  const int waveId = blockIdx.x * 8 + wib;
  const int nWaves = gridDim.x * 8;
  const int nChunk = kNW / 512;           // 864, exact

  v8f cAllP = {}; v8f cAndP = {}; v8f cAllG = {}; v8f cAndG = {};
  v16h ones;
  #pragma unroll
  for (int j = 0; j < 16; ++j) ones[j] = (_Float16)1.0f;

  const uint32_t ldsBase = (uint32_t)(uintptr_t)(&smem[wib][0]);

  for (int ch = waveId; ch < nChunk; ch += nWaves) {   // wave-uniform loop
    const int  base = ch * 512;
    const bool pred = (base < 2 * kNWB);               // chunks never straddle
    const int  moff = pred ? (2 * kNWB) : (-2 * kNWB);

    const uint32_t* gs = skel + base;
    const uint32_t* gm = bits + base + moff;

    // Async-stage 2KB skel + 2KB mask into this wave's LDS slice (ASYNCcnt).
    #pragma unroll
    for (int t = 0; t < 4; ++t) {
      uint64_t ga = (uint64_t)(uintptr_t)gs + (uint64_t)(t * 512 + lane * 16);
      uint32_t la = ldsBase + (uint32_t)(t * 512 + lane * 16);
      asm volatile("global_load_async_to_lds_b128 %0, %1, off"
                   :: "v"(la), "v"(ga) : "memory");
    }
    #pragma unroll
    for (int t = 0; t < 4; ++t) {
      uint64_t ga = (uint64_t)(uintptr_t)gm + (uint64_t)(t * 512 + lane * 16);
      uint32_t la = ldsBase + 2048u + (uint32_t)(t * 512 + lane * 16);
      asm volatile("global_load_async_to_lds_b128 %0, %1, off"
                   :: "v"(la), "v"(ga) : "memory");
    }
    asm volatile("s_wait_asynccnt 0x0" ::: "memory");

    const uint32_t* ss = &smem[wib][0];
    v16h aAll, aAnd;
    #pragma unroll
    for (int j = 0; j < 16; ++j) {
      uint32_t s = ss[lane * 16 + j];
      uint32_t m = ss[512 + lane * 16 + j];
      aAll[j] = (_Float16)(float)__popc(s);
      aAnd[j] = (_Float16)(float)__popc(s & m);
    }
    if (pred) {   // wave-uniform branch: EXEC all-ones (or block skipped) at WMMA
      cAllP = __builtin_amdgcn_wmma_f32_16x16x32_f16(false, aAll, false, ones,
                                                     (short)0, cAllP, false, false);
      cAndP = __builtin_amdgcn_wmma_f32_16x16x32_f16(false, aAnd, false, ones,
                                                     (short)0, cAndP, false, false);
    } else {
      cAllG = __builtin_amdgcn_wmma_f32_16x16x32_f16(false, aAll, false, ones,
                                                     (short)0, cAllG, false, false);
      cAndG = __builtin_amdgcn_wmma_f32_16x16x32_f16(false, aAnd, false, ones,
                                                     (short)0, cAndG, false, false);
    }
  }

  float t0 = 0.f, t1 = 0.f, t2 = 0.f, t3 = 0.f;
  #pragma unroll
  for (int rr = 0; rr < 8; ++rr) {
    t0 += cAndP[rr]; t1 += cAllP[rr]; t2 += cAndG[rr]; t3 += cAllG[rr];
  }
  // lanes 0-15 hold rows 0-7, lanes 16-31 hold rows 8-15 (all columns equal)
  t0 += __shfl_xor(t0, 16, 32);
  t1 += __shfl_xor(t1, 16, 32);
  t2 += __shfl_xor(t2, 16, 32);
  t3 += __shfl_xor(t3, 16, 32);
  if (lane == 0) {
    atomicAdd(acc + 0, t0);
    atomicAdd(acc + 1, t1);
    atomicAdd(acc + 2, t2);
    atomicAdd(acc + 3, t3);
  }
}

__global__ void k_zero_acc(float* acc) {
  int i = threadIdx.x;
  if (i < 4) acc[i] = 0.0f;
}

__global__ void k_final(const float* __restrict__ acc, float* __restrict__ out) {
  float sPY = acc[0], sP = acc[1], sGX = acc[2], sG = acc[3];
  float tprec = (sPY + 1.0f) / (sP + 1.0f);
  float tsens = (sGX + 1.0f) / (sG + 1.0f);
  float cl = 2.0f * (tprec * tsens + 1.0f) / (tprec + tsens + 1.0f);
  out[0] = -cl;
}

// ---------------------------------------------------------------------------
// Host-side pipeline (graph-capture safe: kernel launches only)
// ---------------------------------------------------------------------------
extern "C" void kernel_launch(void* const* d_in, const int* in_sizes, int n_in,
                              void* d_out, int out_size, void* d_ws, size_t ws_size,
                              hipStream_t stream) {
  (void)in_sizes; (void)n_in; (void)out_size; (void)ws_size;
  const float* x = (const float*)d_in[0];
  const float* y = (const float*)d_in[1];
  float* out = (float*)d_out;

  uint32_t* ws   = (uint32_t*)d_ws;
  uint32_t* BITS = ws + 0 * (size_t)kNW;   // packed x_all|y_all reference masks
  uint32_t* A    = ws + 1 * (size_t)kNW;   // current eroded x
  uint32_t* B    = ws + 2 * (size_t)kNW;   // next x
  uint32_t* E    = ws + 3 * (size_t)kNW;   // erosion temp (for opening)
  uint32_t* SK   = ws + 4 * (size_t)kNW;   // skeleton
  float*    ACC  = (float*)(ws + 5 * (size_t)kNW);

  const int nt  = 256;
  const int nbW = (kNW + nt - 1) / nt;     // 1728 blocks (word-grain)
  const int nbR = (kNRow + nt - 1) / nt;   // 288 blocks (row-grain)

  k_binarize<<<nbW, nt, 0, stream>>>(x, y, BITS, A);

  // s = x & ~open(x)
  k_erode_row<<<nbR, nt, 0, stream>>>(A, E);
  k_dilate_skel_row<<<nbR, nt, 0, stream>>>(E, A, SK, 1);

  for (int it = 0; it < kIters; ++it) {
    k_erode_row<<<nbR, nt, 0, stream>>>(A, B);               // x = erode(x)
    k_erode_row<<<nbR, nt, 0, stream>>>(B, E);               // erode for opening
    k_dilate_skel_row<<<nbR, nt, 0, stream>>>(E, B, SK, 0);  // skel |= x & ~open(x)
    uint32_t* t = A; A = B; B = t;
  }

  k_zero_acc<<<1, 32, 0, stream>>>(ACC);
  k_reduce<<<108, 256, 0, stream>>>(SK, BITS, ACC);   // 108*8 waves == 864 chunks
  k_final<<<1, 1, 0, stream>>>(ACC, out);
}
